// ISTFT_70866960384701
// MI455X (gfx1250) — matrix operand order repaired and checked by reading, compiled, not run on hardware
//
#include <hip/hip_runtime.h>

#define WIN      2048
#define HOP      512
#define NFRAMES  16384
#define NBINS    1025
#define KDIM     2048          // packed K: 1025 cos rows + 1023 sin rows
#define KCH      64            // KDIM / 32
#define NTILES   128           // WIN / 16
#define SIG_LEN  (NFRAMES * HOP + WIN)   // 8390656
#define OLA_SPAN (15 * HOP + WIN)        // 9728 floats per block
#define TRIM     (3 * HOP)

typedef __attribute__((ext_vector_type(16))) __bf16 v16bf;
typedef __attribute__((ext_vector_type(8)))  float  v8f;

static __device__ __forceinline__ unsigned short f2bf(float f) {
  unsigned u = __float_as_uint(f);
  u += 0x7FFFu + ((u >> 16) & 1u);          // round-to-nearest-even
  return (unsigned short)(u >> 16);
}

// ---------------- synthesis window rs[j] (Griffin-Lim ideal window) ----------
__global__ void k_build_rs(float* __restrict__ rs) {
  int j = blockIdx.x * blockDim.x + threadIdx.x;
  if (j >= WIN) return;
  const float twopi = 6.2831853071795864f;
  float wj = 0.54f - 0.46f * cosf(twopi * (float)j / 2047.0f);   // np.hamming
  float env = 0.0f;
  for (int k = -4; k <= 4; ++k) {
    int idx = j - HOP * k;
    if (idx >= 0 && idx < WIN) {
      float w = 0.54f - 0.46f * cosf(twopi * (float)idx / 2047.0f);
      env += w * w * (1.0f / 4094.0f);      // fft_size = (WIN-1)*2 = 4094
    }
  }
  rs[j] = wj / (sqrtf(4094.0f) * env);      // syn_w / env
}

// ---------------- DFT basis, stored directly in WMMA B-fragment order --------
// row k<1025:  c_k * cos(2*pi*k*n2/2048),  n2 = (j+1024) mod 2048 (fftshift folded)
// row k>=1025: kr=k-1024 in [1,1023]:  -(2/2048) * sin(2*pi*kr*n2/2048)
__global__ void k_build_basis(unsigned short* __restrict__ B) {
  int idx = blockIdx.x * blockDim.x + threadIdx.x;
  if (idx >= KDIM * WIN) return;
  int k = idx >> 11;             // / 2048
  int n = idx & (WIN - 1);
  int kr = (k < NBINS) ? k : (k - 1024);
  int n2 = (n + WIN / 2) & (WIN - 1);
  int phint = (kr * n2) & (WIN - 1);                 // exact mod-2048 reduction
  float ang = 0.0030679615757712823f * (float)phint; // 2*pi/2048
  float c = ((kr == 0) || (kr == NBINS - 1)) ? (1.0f / 2048.0f) : (2.0f / 2048.0f);
  float val = (k < NBINS) ? (c * cosf(ang)) : (-c * sinf(ang));
  // B-fragment order used here (contiguous k): lanes 0-15 hold K=0..15, lanes 16-31 hold K=16..31
  int kc = k >> 5, kk = k & 31;
  int nt = n >> 4, nn = n & 15;
  int lane = nn + ((kk & 16) ? 16 : 0);
  int slot = kk & 15;
  B[(((size_t)nt * KCH + kc) * 32 + lane) * 16 + slot] = f2bf(val);
}

// ---------------- main kernel: DFT-as-GEMM (WMMA bf16) + windowed OLA --------
__global__ __launch_bounds__(256) void k_dft_ola(
    const float* __restrict__ mag, const float* __restrict__ ph,
    const unsigned short* __restrict__ Bfrag, const float* __restrict__ rs,
    float* __restrict__ sig)
{
  // smem reused: phase 1 = A-fragment staging (32 chunks * 32 lanes * 32B = 32KB)
  //              phase 2 = fp32 overlap-add buffer (9728 floats = 38912B)
  __shared__ __align__(32) unsigned char smem[OLA_SPAN * 4];
  unsigned short* ldsA = (unsigned short*)smem;
  float*          ola  = (float*)smem;

  const int tid  = threadIdx.x;
  const int t0   = blockIdx.x * 16;          // first frame of this M-tile
  const int wave = tid >> 5;
  const int lane = tid & 31;
  const int N    = lane & 15;
  const int Moff = (lane >> 4) << 3;         // C/D layout: lanes 16-31 -> M += 8

  v8f acc[16] = {};                          // 16 N-tiles per wave, fp32 accum

  const unsigned short* bw = Bfrag + (size_t)wave * 16 * (KCH * 512) + (size_t)lane * 16;

  for (int h = 0; h < 2; ++h) {
    __syncthreads();
    // ---- stage A fragments for K chunks [h*32, h*32+32), contiguous-k lane/slot order (not ISA table order)
    for (int li = tid; li < 16 * 1024; li += 256) {
      int m  = li >> 10;
      int kl = li & 1023;
      int k  = h * 1024 + kl;                // global packed-K index
      int kr = (k < NBINS) ? k : (k - 1024);
      float mg = mag[(size_t)(t0 + m) * NBINS + kr];
      float p  = ph [(size_t)(t0 + m) * NBINS + kr];
      float v  = (k < NBINS) ? (mg * cosf(p)) : (mg * sinf(p));
      // A order used here (contiguous k): lanes 0-15 hold K=0..15, lanes 16-31 hold K=16..31 (matches B)
      int kc_l = kl >> 5, kk = k & 31;
      int alane = m + ((kk & 16) ? 16 : 0);
      int slot  = kk & 15;
      ldsA[(kc_l * 32 + alane) * 16 + slot] = f2bf(v);
    }
    __syncthreads();
    // ---- GEMM: 32 K-chunks, 16 N-tiles
    for (int kc = 0; kc < 32; ++kc) {
      int kcg = h * 32 + kc;
      v16bf a = *reinterpret_cast<const v16bf*>(&ldsA[(kc * 32 + lane) * 16]);
      #pragma unroll
      for (int i = 0; i < 16; ++i) {
        const v16bf* bp = reinterpret_cast<const v16bf*>(
            bw + (size_t)i * (KCH * 512) + (size_t)kcg * 512);
        acc[i] = __builtin_amdgcn_wmma_f32_16x16x32_bf16(
            false, a, false, *bp, (short)0, acc[i], false, false);
      }
    }
  }
  __syncthreads();

  // ---- zero block-local OLA buffer
  for (int p = tid; p < OLA_SPAN; p += 256) ola[p] = 0.0f;
  __syncthreads();

  // ---- epilogue: apply rs in fp32, fold 128 tiles into LDS (ds_add_f32)
  #pragma unroll
  for (int i = 0; i < 16; ++i) {
    int j = (wave * 16 + i) * 16 + N;        // sample index within frame
    float rsv = rs[j];
    #pragma unroll
    for (int r = 0; r < 8; ++r) {
      atomicAdd(&ola[(Moff + r) * HOP + j], acc[i][r] * rsv);
    }
  }
  __syncthreads();

  // ---- flush to global signal; adjacent blocks overlap by 1536 -> atomics
  const size_t base = (size_t)t0 * HOP;
  for (int p = tid; p < OLA_SPAN; p += 256) {
    atomicAdd(&sig[base + p], ola[p]);
  }
}

// ---------------- trim transients ---------------------------------------
__global__ void k_trim(const float* __restrict__ sig, float* __restrict__ out, int n) {
  int i = blockIdx.x * blockDim.x + threadIdx.x;
  if (i < n) out[i] = sig[i + TRIM];
}

extern "C" void kernel_launch(void* const* d_in, const int* in_sizes, int n_in,
                              void* d_out, int out_size, void* d_ws, size_t ws_size,
                              hipStream_t stream) {
  const float* mag = (const float*)d_in[0];
  const float* ph  = (const float*)d_in[1];
  float* out = (float*)d_out;

  // workspace layout (all offsets 32B aligned)
  float*          sig   = (float*)d_ws;                    // SIG_LEN floats
  float*          rs    = sig + SIG_LEN;                   // 2048 floats
  unsigned short* Bfrag = (unsigned short*)(rs + WIN);     // KDIM*WIN bf16

  hipMemsetAsync(sig, 0, (size_t)SIG_LEN * sizeof(float), stream);
  k_build_rs<<<WIN / 256, 256, 0, stream>>>(rs);
  k_build_basis<<<(KDIM * WIN) / 256, 256, 0, stream>>>(Bfrag);
  k_dft_ola<<<NFRAMES / 16, 256, 0, stream>>>(mag, ph, Bfrag, rs, sig);
  k_trim<<<(out_size + 255) / 256, 256, 0, stream>>>(sig, out, out_size);
}
